// AttentionBlock_25374666785241
// MI455X (gfx1250) — matrix operand order repaired and checked by reading, not compiled, not run on hardware
//
#include <hip/hip_runtime.h>
#include <hip/hip_bf16.h>
#include <math.h>

#define TPB 256

#if __has_builtin(__builtin_amdgcn_global_load_async_to_lds_b128)
#define HAVE_ASYNC_LDS 1
#endif

typedef __bf16 bf16x16 __attribute__((ext_vector_type(16)));
typedef float f32x8 __attribute__((ext_vector_type(8)));
typedef int v4i __attribute__((ext_vector_type(4)));

union FragU {
  uint4 u4[2];
  bf16x16 bf;
};

__device__ __forceinline__ unsigned short f2bf(float f) {
  unsigned int u = __float_as_uint(f);
  u += 0x7FFFu + ((u >> 16) & 1u);  // round-to-nearest-even
  return (unsigned short)(u >> 16);
}
__device__ __forceinline__ float bf2f(unsigned short h) {
  return __uint_as_float(((unsigned int)h) << 16);
}

// Async 16-byte global -> LDS copy (ASYNCcnt path); fallback: load + ds_store.
__device__ __forceinline__ void async_b128(const unsigned short* g,
                                           unsigned short* l) {
#if defined(HAVE_ASYNC_LDS)
  __builtin_amdgcn_global_load_async_to_lds_b128(
      (__attribute__((address_space(1))) v4i*)g,
      (__attribute__((address_space(3))) v4i*)l, 0, 0);
#else
  *reinterpret_cast<uint4*>(l) = *reinterpret_cast<const uint4*>(g);
#endif
}

__device__ __forceinline__ void wait_async_all() {
#if defined(HAVE_ASYNC_LDS)
#if __has_builtin(__builtin_amdgcn_s_wait_asynccnt)
  __builtin_amdgcn_s_wait_asynccnt(0);
#else
  asm volatile("s_wait_asynccnt 0" ::: "memory");
#endif
#endif
}

// ---------------------------------------------------------------------------
// fp32 -> bf16 conversion (weights)
// ---------------------------------------------------------------------------
__global__ void cvt_f32_bf16(const float* __restrict__ in,
                             unsigned short* __restrict__ out, int n) {
  int i = blockIdx.x * blockDim.x + threadIdx.x;
  if (i < n) out[i] = f2bf(in[i]);
}

// ---------------------------------------------------------------------------
// GroupNorm over [B, C, N]; writes bf16 token-major hT[B, N, C].
// One block per (group, batch). C/G = 16 channels, N = 1024 pixels.
// ---------------------------------------------------------------------------
__global__ __launch_bounds__(TPB)
void groupnorm_bf16(const float* __restrict__ x,
                    const float* __restrict__ gamma,
                    const float* __restrict__ beta,
                    unsigned short* __restrict__ hT,
                    int C, int Npix, int Cpg) {
  const int g = blockIdx.x;
  const int b = blockIdx.y;
  const int tid = threadIdx.x;
  const float* xb = x + ((size_t)b * C + (size_t)g * Cpg) * Npix;

  float sum = 0.f, sq = 0.f;
  for (int c = 0; c < Cpg; ++c) {
    for (int n = tid; n < Npix; n += TPB) {
      float t = xb[(size_t)c * Npix + n];
      sum += t;
      sq += t * t;
    }
  }
  __shared__ float rs[TPB], rq[TPB];
  rs[tid] = sum; rq[tid] = sq;
  __syncthreads();
  for (int off = TPB / 2; off > 0; off >>= 1) {
    if (tid < off) { rs[tid] += rs[tid + off]; rq[tid] += rq[tid + off]; }
    __syncthreads();
  }
  const float inv_cnt = 1.0f / (float)(Cpg * Npix);
  const float mean = rs[0] * inv_cnt;
  const float var = rq[0] * inv_cnt - mean * mean;
  const float rstd = rsqrtf(var + 1e-5f);

  for (int n = tid; n < Npix; n += TPB) {
    unsigned short* dst = hT + ((size_t)b * Npix + n) * C + (size_t)g * Cpg;
    for (int c = 0; c < Cpg; ++c) {
      float t = (xb[(size_t)c * Npix + n] - mean) * rstd;
      t = t * gamma[g * Cpg + c] + beta[g * Cpg + c];
      dst[c] = f2bf(t);
    }
  }
}

// ---------------------------------------------------------------------------
// In-place bf16 row softmax: one block per row of length Nn (=1024).
// ---------------------------------------------------------------------------
__global__ __launch_bounds__(TPB)
void softmax_rows(unsigned short* __restrict__ attn, int Nn) {
  const int tid = threadIdx.x;
  unsigned short* row =
      attn + ((size_t)blockIdx.y * gridDim.x + blockIdx.x) * Nn;
  const int PER = 4;  // Nn / TPB

  float v[PER];
  float m = -1e30f;
#pragma unroll
  for (int i = 0; i < PER; ++i) {
    v[i] = bf2f(row[tid + i * TPB]);
    m = fmaxf(m, v[i]);
  }
  __shared__ float red[TPB];
  red[tid] = m;
  __syncthreads();
  for (int off = TPB / 2; off > 0; off >>= 1) {
    if (tid < off) red[tid] = fmaxf(red[tid], red[tid + off]);
    __syncthreads();
  }
  const float rowmax = red[0];
  __syncthreads();

  float s = 0.f;
#pragma unroll
  for (int i = 0; i < PER; ++i) {
    v[i] = __expf(v[i] - rowmax);
    s += v[i];
  }
  red[tid] = s;
  __syncthreads();
  for (int off = TPB / 2; off > 0; off >>= 1) {
    if (tid < off) red[tid] += red[tid + off];
    __syncthreads();
  }
  const float inv = 1.0f / red[0];
#pragma unroll
  for (int i = 0; i < PER; ++i) row[tid + i * TPB] = f2bf(v[i] * inv);
}

// ---------------------------------------------------------------------------
// Tiled bf16 WMMA GEMM with double-buffered async LDS staging:
//   D[M x Nn] = A[M x K] * Bt[Nn x K]^T  (+ epilogue)
// Block = 8 waves -> 128x128 tile; wave -> 32x64 (2x4 wmma accumulators).
// Per k-step a block stages A(128x32) and B(128x32) bf16 tiles into LDS via
// GLOBAL_LOAD_ASYNC_TO_LDS_B128 (ASYNCcnt) while computing the previous tile
// from LDS (ds_load_b128 fragment reads, rows padded to 80B: conflict-free).
// MODE 0: bf16 out row-major, + bias[col]              (Q, K projections)
// MODE 1: bf16 out row-major, * scale                  (scores, attn@V)
// MODE 2: bf16 out col-major [col*M + row], + bias     (V projection)
// MODE 4: f32  out col-major, + bias + resid, d_out    (out proj + residual)
// ---------------------------------------------------------------------------
#define LDS_STRIDE 40   // ushorts per staged row (32 data + 8 pad = 80B)
#define TILE_USH (128 * LDS_STRIDE)   // 5120 ushorts per tile
#define BUF_USH (2 * TILE_USH)        // A + B per buffer

template <int MODE>
__global__ __launch_bounds__(TPB)
void gemm_bf16_wmma(const unsigned short* __restrict__ A, long long aStride,
                    const unsigned short* __restrict__ Bt, long long bStride,
                    int M, int Nn, int K,
                    const float* __restrict__ bias,
                    const float* __restrict__ resid, long long rStride,
                    float* __restrict__ outF,
                    unsigned short* __restrict__ outH, long long oStride,
                    float scale) {
  __shared__ unsigned short smem[2 * BUF_USH];  // 40 KB

  const int b = blockIdx.z;
  A += (size_t)b * aStride;
  Bt += (size_t)b * bStride;

  const int tid = threadIdx.x;
  const int wave = tid >> 5;
  const int lane = tid & 31;
  const int lo = lane & 15;
  const int hi = lane >> 4;

  const int rowBlk = blockIdx.y * 128;
  const int colBlk = blockIdx.x * 128;
  const int waveRow = (wave >> 1) * 32;
  const int waveCol = (wave & 1) * 64;

  const int crow = tid >> 2;  // 0..63: staged row handled by this thread
  const int cseg = tid & 3;   // 16B segment within the 64B row

  f32x8 acc[2][4] = {};
  const int nK = K >> 5;

  auto stage = [&](int buf, int k0) {
    unsigned short* aD = smem + buf * BUF_USH;
    unsigned short* bD = aD + TILE_USH;
#pragma unroll
    for (int p = 0; p < 2; ++p) {
      const int row = crow + p * 64;
      async_b128(A + (size_t)(rowBlk + row) * K + k0 + cseg * 8,
                 aD + row * LDS_STRIDE + cseg * 8);
      async_b128(Bt + (size_t)(colBlk + row) * K + k0 + cseg * 8,
                 bD + row * LDS_STRIDE + cseg * 8);
    }
    if (k0 + 32 < K) {
      __builtin_prefetch(A + (size_t)(rowBlk + crow) * K + k0 + 32, 0, 1);
      __builtin_prefetch(Bt + (size_t)(colBlk + crow) * K + k0 + 32, 0, 1);
    }
  };

  stage(0, 0);
  wait_async_all();
  __syncthreads();

  for (int kt = 0; kt < nK; ++kt) {
    const int cur = kt & 1;
    if (kt + 1 < nK) stage(cur ^ 1, (kt + 1) << 5);

    const unsigned short* aB = smem + cur * BUF_USH;
    const unsigned short* bB = aB + TILE_USH;

    FragU af[2], bf[4];
    // A fragment (16x32): lane lo = row; K chunks at hi*8 and 16+hi*8
#pragma unroll
    for (int mi = 0; mi < 2; ++mi) {
      const unsigned short* p =
          aB + (waveRow + mi * 16 + lo) * LDS_STRIDE + hi * 8;
      af[mi].u4[0] = *reinterpret_cast<const uint4*>(p);
      af[mi].u4[1] = *reinterpret_cast<const uint4*>(p + 16);
    }
    // B fragment (32x16): lane lo = col; K chunks at hi*8 and 16+hi*8
#pragma unroll
    for (int ni = 0; ni < 4; ++ni) {
      const uint4* p = reinterpret_cast<const uint4*>(
          bB + (waveCol + ni * 16 + lo) * LDS_STRIDE + hi * 8);
      bf[ni].u4[0] = p[0];
      bf[ni].u4[1] = p[2];
    }
#pragma unroll
    for (int mi = 0; mi < 2; ++mi) {
#pragma unroll
      for (int ni = 0; ni < 4; ++ni) {
        acc[mi][ni] = __builtin_amdgcn_wmma_f32_16x16x32_bf16(
            false, af[mi].bf, false, bf[ni].bf, (short)0, acc[mi][ni], false,
            false);
      }
    }

    wait_async_all();
    __syncthreads();
  }

  // Epilogue. D layout: VGPR r -> row hi*8 + r, col = lo.
  const int Npix = M;
#pragma unroll
  for (int mi = 0; mi < 2; ++mi) {
#pragma unroll
    for (int ni = 0; ni < 4; ++ni) {
      const int col = colBlk + waveCol + ni * 16 + lo;
      const int row0 = rowBlk + waveRow + mi * 16 + hi * 8;
      f32x8 d = acc[mi][ni];
      if (MODE == 0) {
        const float bv = bias[col];
        unsigned short* o = outH + (size_t)b * oStride;
#pragma unroll
        for (int r = 0; r < 8; ++r)
          o[(size_t)(row0 + r) * Nn + col] = f2bf(d[r] + bv);
      } else if (MODE == 1) {
        unsigned short* o = outH + (size_t)b * oStride;
#pragma unroll
        for (int r = 0; r < 8; ++r)
          o[(size_t)(row0 + r) * Nn + col] = f2bf(d[r] * scale);
      } else if (MODE == 2) {
        const float bv = bias[col];
        unsigned short* o = outH + (size_t)b * oStride;
#pragma unroll
        for (int r = 0; r < 8; ++r)
          o[(size_t)col * Npix + row0 + r] = f2bf(d[r] + bv);
      } else {  // MODE == 4
        const float bv = bias[col];
        float* o = outF + (size_t)b * oStride;
        const float* rx = resid + (size_t)b * rStride;
#pragma unroll
        for (int r = 0; r < 8; ++r) {
          size_t idx = (size_t)col * Npix + row0 + r;
          o[idx] = d[r] + bv + rx[idx];
        }
      }
    }
  }
}

// ---------------------------------------------------------------------------
// Host-side orchestration
// ---------------------------------------------------------------------------
extern "C" void kernel_launch(void* const* d_in, const int* in_sizes, int n_in,
                              void* d_out, int out_size, void* d_ws,
                              size_t ws_size, hipStream_t stream) {
  (void)in_sizes; (void)n_in; (void)out_size; (void)ws_size;

  const int B = 16, C = 512, NPIX = 1024;  // H*W = 32*32
  const int G = 32, CPG = C / G;
  const float scale = 1.0f / sqrtf((float)C);

  const float* x   = (const float*)d_in[0];
  const float* gnw = (const float*)d_in[1];
  const float* gnb = (const float*)d_in[2];
  const float* wq  = (const float*)d_in[3];
  const float* bq  = (const float*)d_in[4];
  const float* wk  = (const float*)d_in[5];
  const float* bk  = (const float*)d_in[6];
  const float* wv  = (const float*)d_in[7];
  const float* bv  = (const float*)d_in[8];
  const float* wo  = (const float*)d_in[9];
  const float* bo  = (const float*)d_in[10];
  float* out = (float*)d_out;

  // Workspace carve-up (bf16 everywhere, ~114 MB total)
  char* ws = (char*)d_ws;
  size_t off = 0;
  auto alloc = [&](size_t bytes) -> char* {
    char* p = ws + off;
    off += (bytes + 255) & ~(size_t)255;
    return p;
  };
  const size_t NC = (size_t)NPIX * C;
  const size_t NN = (size_t)NPIX * NPIX;
  unsigned short* hT   = (unsigned short*)alloc((size_t)B * NC * 2);  // [B,N,C]
  unsigned short* qB   = (unsigned short*)alloc((size_t)B * NC * 2);  // [B,N,C]
  unsigned short* kT   = (unsigned short*)alloc((size_t)B * NC * 2);  // [B,N,C]
  unsigned short* vCM  = (unsigned short*)alloc((size_t)B * NC * 2);  // [B,C,N]
  unsigned short* oMid = (unsigned short*)alloc((size_t)B * NC * 2);  // [B,N,C]
  unsigned short* attn = (unsigned short*)alloc((size_t)B * NN * 2);  // [B,N,N]
  unsigned short* wqb  = (unsigned short*)alloc((size_t)C * C * 2);
  unsigned short* wkb  = (unsigned short*)alloc((size_t)C * C * 2);
  unsigned short* wvb  = (unsigned short*)alloc((size_t)C * C * 2);
  unsigned short* wob  = (unsigned short*)alloc((size_t)C * C * 2);

  // 1) weights -> bf16
  {
    const int n = C * C;
    dim3 g((n + TPB - 1) / TPB);
    cvt_f32_bf16<<<g, TPB, 0, stream>>>(wq, wqb, n);
    cvt_f32_bf16<<<g, TPB, 0, stream>>>(wk, wkb, n);
    cvt_f32_bf16<<<g, TPB, 0, stream>>>(wv, wvb, n);
    cvt_f32_bf16<<<g, TPB, 0, stream>>>(wo, wob, n);
  }

  // 2) GroupNorm -> hT (bf16, token-major)
  groupnorm_bf16<<<dim3(G, B), TPB, 0, stream>>>(x, gnw, gnb, hT, C, NPIX, CPG);

  // 3) Q/K/V projections (M=1024, Nn=512, K=512)
  {
    dim3 g(C / 128, NPIX / 128, B);
    gemm_bf16_wmma<0><<<g, TPB, 0, stream>>>(hT, (long long)NC, wqb, 0, NPIX, C,
                                             C, bq, nullptr, 0, nullptr, qB,
                                             (long long)NC, 1.0f);
    gemm_bf16_wmma<0><<<g, TPB, 0, stream>>>(hT, (long long)NC, wkb, 0, NPIX, C,
                                             C, bk, nullptr, 0, nullptr, kT,
                                             (long long)NC, 1.0f);
    gemm_bf16_wmma<2><<<g, TPB, 0, stream>>>(hT, (long long)NC, wvb, 0, NPIX, C,
                                             C, bv, nullptr, 0, nullptr, vCM,
                                             (long long)NC, 1.0f);
  }

  // 4) scores = (q @ kT^T) * 1/sqrt(C)   (M=1024, Nn=1024, K=512)
  {
    dim3 g(NPIX / 128, NPIX / 128, B);
    gemm_bf16_wmma<1><<<g, TPB, 0, stream>>>(qB, (long long)NC, kT,
                                             (long long)NC, NPIX, NPIX, C,
                                             nullptr, nullptr, 0, nullptr, attn,
                                             (long long)NN, scale);
  }

  // 5) softmax rows, in place
  softmax_rows<<<dim3(NPIX, B), TPB, 0, stream>>>(attn, NPIX);

  // 6) o = attn @ v   (M=1024, Nn=512, K=1024); vCM rows = channels (K-contig)
  {
    dim3 g(C / 128, NPIX / 128, B);
    gemm_bf16_wmma<1><<<g, TPB, 0, stream>>>(attn, (long long)NN, vCM,
                                             (long long)NC, NPIX, C, NPIX,
                                             nullptr, nullptr, 0, nullptr, oMid,
                                             (long long)NC, 1.0f);
  }

  // 7) out = x + (o @ wo^T + bo), stored fp32 as [B, C, N]
  {
    dim3 g(C / 128, NPIX / 128, B);
    gemm_bf16_wmma<4><<<g, TPB, 0, stream>>>(oMid, (long long)NC, wob, 0, NPIX,
                                             C, C, bo, x, (long long)NC, out,
                                             nullptr, (long long)NC, 1.0f);
  }
}
